// DynamicSlimmableBlock_40510131535959
// MI455X (gfx1250) — matrix operand order repaired and checked by reading, not compiled, not run on hardware
//
#include <hip/hip_runtime.h>

// ---------------------------------------------------------------------------
// DynamicSlimmableBlock for MI455X (gfx1250, wave32, WMMA).
// Fusion: out = M + depthwise_conv(M), where
//   M[b,t,o] = (p0+p1)*sum_{h<256} relu(x@w1^T+b1)_h * w2[o,h]
//            +  p0    *sum_{256<=h<2048} relu(...)_h * w2[o,h]
// GEMMs use v_wmma_f32_16x16x32_bf16 (bf16 in, f32 accumulate).
// ---------------------------------------------------------------------------

typedef __attribute__((ext_vector_type(16))) __bf16 v16bf;
typedef __attribute__((ext_vector_type(8)))  float  v8f;

#define B_SZ 8
#define T_SZ 4000
#define CIN 256
#define COUT 256
#define HID 2048
#define KLEN 39
#define PAD 19
#define ROWS (B_SZ * T_SZ)          // 32000, divisible by 16; 250 tiles per batch
#define TILES (ROWS / 16)           // 2000
#define TILES_PER_B (T_SZ / 16)     // 250
#define LDSW 272                    // 256 + 16 bf16 pad to dodge bank conflicts
#define OUT_N (ROWS * COUT)         // 8192000

union AF { unsigned int u[8]; v16bf v; };

__device__ __forceinline__ __bf16 f2bf(float f) {
  unsigned int u = __builtin_bit_cast(unsigned int, f);
  u += 0x7FFFu + ((u >> 16) & 1u);                 // round-to-nearest-even
  unsigned short h = (unsigned short)(u >> 16);
  return __builtin_bit_cast(__bf16, h);
}

// 16-bit A-matrix 16x32 fragment in contiguous k order, matching the packed B:
// lane: m = lane&15, half = lane>>4; VGPR v holds K pair at
// k = half*16 + v*2   (+k0 for the K-step)
__device__ __forceinline__ v16bf lds_afrag(const __bf16* S, int lane, int k0) {
  const int m = lane & 15, lh = lane >> 4;
  const int base = m * LDSW + k0 + lh * 16;
  AF f;
#pragma unroll
  for (int vv = 0; vv < 8; ++vv) {
    const int k = vv * 2;
    f.u[vv] = *(const unsigned int*)(S + base + k);
  }
  return f.v;
}

// -------------------------- gate: mean/std -> softmax ----------------------
__global__ __launch_bounds__(256) void gate_kernel(
    const float* __restrict__ x, const float* __restrict__ gw,
    const float* __restrict__ gb, const float* __restrict__ tau,
    float* __restrict__ probs) {
  const int b = blockIdx.x, c = threadIdx.x;
  const float* xb = x + (size_t)b * T_SZ * CIN + c;
  float s = 0.f, sq = 0.f;
  for (int t = 0; t < T_SZ; ++t) { float v = xb[(size_t)t * CIN]; s += v; sq += v * v; }
  const float mean = s * (1.f / T_SZ);
  const float var  = (sq - (float)T_SZ * mean * mean) * (1.f / (T_SZ - 1));
  const float sd   = sqrtf(fmaxf(var, 0.f));
  float p0 = mean * gw[c]       + sd * gw[CIN + c];          // gate_w[0,:]
  float p1 = mean * gw[512 + c] + sd * gw[512 + CIN + c];    // gate_w[1,:]
  __shared__ float r0[256], r1[256];
  r0[c] = p0; r1[c] = p1; __syncthreads();
  for (int sft = 128; sft > 0; sft >>= 1) {
    if (c < sft) { r0[c] += r0[c + sft]; r1[c] += r1[c + sft]; }
    __syncthreads();
  }
  if (c == 0) {
    const float tv = tau[0];
    const float l0 = tv * (r0[0] + gb[0]);
    const float l1 = tv * (r1[0] + gb[1]);
    const float mx = fmaxf(l0, l1);
    const float e0 = __expf(l0 - mx), e1 = __expf(l1 - mx);
    const float inv = 1.f / (e0 + e1);
    probs[b * 2 + 0] = e0 * inv;
    probs[b * 2 + 1] = e1 * inv;
  }
}

// -------------------------- scalar aux losses ------------------------------
__global__ void loss_kernel(const float* __restrict__ probs, float* __restrict__ outl) {
  if (threadIdx.x == 0 && blockIdx.x == 0) {
    float sp = 0.f, i0 = 0.f, i1 = 0.f;
    for (int b = 0; b < B_SZ; ++b) {
      const float p0 = probs[2 * b], p1 = probs[2 * b + 1];
      const float nrm = sqrtf(p0 * p0 + p1 * p1) + 1e-8f;
      sp += (fabsf(p0) + fabsf(p1)) / nrm;
      i0 += p0; i1 += p1;
    }
    sp = 0.1f * sp * (1.f / B_SZ);                 // ALPHA * mean_b sum_k |np|
    i0 *= (1.f / B_SZ); i1 *= (1.f / B_SZ);
    const float mu = 0.5f * (i0 + i1);
    const float sd = fabsf(i0 - i1) * 0.70710678118f; // ddof=1 for 2 samples
    const float cv = sd / (mu + 1e-8f);
    outl[0] = sp;
    outl[1] = 0.01f * cv * cv;                     // BETA * cv^2
  }
}

// ------------- pack w1 [2048,256] into bf16 WMMA B-fragment order ----------
// index = ((ht*8 + ks)*32 + lane)*16 + elem; value = w1[ht*16 + lane%16,
//          ks*32 + (lane/16)*16 + elem]
__global__ __launch_bounds__(256) void pack_w1_kernel(
    const float* __restrict__ w1, __bf16* __restrict__ w1b) {
  const int gid = blockIdx.x * 256 + threadIdx.x;      // [0, 2048*256)
  const int elem = gid & 15;
  const int lane = (gid >> 4) & 31;
  const int ks   = (gid >> 9) & 7;
  const int ht   = gid >> 12;                          // 0..127
  const int h = ht * 16 + (lane & 15);
  const int c = ks * 32 + (lane >> 4) * 16 + elem;
  w1b[gid] = f2bf(w1[h * CIN + c]);
}

// ------------- pack w2 [256,2048] into bf16 WMMA B-fragment order ----------
// index = ((ot*64 + kg)*32 + lane)*16 + elem; value = w2[ot*16 + lane%16,
//          kg*32 + (lane/16)*16 + elem]
__global__ __launch_bounds__(256) void pack_w2_kernel(
    const float* __restrict__ w2, __bf16* __restrict__ w2b) {
  const int gid = blockIdx.x * 256 + threadIdx.x;      // [0, 256*2048)
  const int elem = gid & 15;
  const int lane = (gid >> 4) & 31;
  const int kg   = (gid >> 9) & 63;
  const int ot   = gid >> 15;                          // 0..15
  const int o = ot * 16 + (lane & 15);
  const int h = kg * 32 + (lane >> 4) * 16 + elem;
  w2b[gid] = f2bf(w2[o * HID + h]);
}

// ---- fused GEMM1 (relu) + GEMM2 (split lo/hi accumulators) + gate blend ---
// One block = 16 rows of (b,t). 8 waves; wave w owns hidden tiles w*2,w*2+1
// of each 256-wide chunk, and output tiles w*2,w*2+1 (32 out channels).
__device__ __forceinline__ void do_chunk(
    int hc, const float* __restrict__ b1,
    const __bf16* __restrict__ w1b, const __bf16* __restrict__ w2b,
    const __bf16* xs, __bf16* fs, int lane, int wid, int nn, int lh,
    v8f& a0, v8f& a1) {
  v8f c0 = {}, c1 = {};
  const int ht0 = hc * 16 + wid * 2;
#pragma unroll
  for (int ks = 0; ks < 8; ++ks) {
    v16bf a = lds_afrag(xs, lane, ks * 32);
    v16bf bm0 = *(const v16bf*)(w1b + ((size_t)((ht0 + 0) * 8 + ks) * 32 + lane) * 16);
    v16bf bm1 = *(const v16bf*)(w1b + ((size_t)((ht0 + 1) * 8 + ks) * 32 + lane) * 16);
    c0 = __builtin_amdgcn_wmma_f32_16x16x32_bf16(false, a, false, bm0, (short)0, c0, false, false);
    c1 = __builtin_amdgcn_wmma_f32_16x16x32_bf16(false, a, false, bm1, (short)0, c1, false, false);
  }
  // bias + relu, write bf16 f1 chunk to LDS (C layout: m = v + 8*half, n = lane&15)
  const int htl0 = wid * 2, htl1 = wid * 2 + 1;
  const float bias0 = b1[(hc * 16 + htl0) * 16 + nn];
  const float bias1 = b1[(hc * 16 + htl1) * 16 + nn];
#pragma unroll
  for (int vv = 0; vv < 8; ++vv) {
    const int m = vv + lh * 8;
    fs[m * LDSW + htl0 * 16 + nn] = f2bf(fmaxf(c0[vv] + bias0, 0.f));
    fs[m * LDSW + htl1 * 16 + nn] = f2bf(fmaxf(c1[vv] + bias1, 0.f));
  }
  __syncthreads();
  const int ot0 = wid * 2;
#pragma unroll
  for (int ks = 0; ks < 8; ++ks) {
    v16bf a = lds_afrag(fs, lane, ks * 32);
    const int kg = hc * 8 + ks;
    v16bf bm0 = *(const v16bf*)(w2b + ((size_t)((ot0 + 0) * 64 + kg) * 32 + lane) * 16);
    v16bf bm1 = *(const v16bf*)(w2b + ((size_t)((ot0 + 1) * 64 + kg) * 32 + lane) * 16);
    a0 = __builtin_amdgcn_wmma_f32_16x16x32_bf16(false, a, false, bm0, (short)0, a0, false, false);
    a1 = __builtin_amdgcn_wmma_f32_16x16x32_bf16(false, a, false, bm1, (short)0, a1, false, false);
  }
  __syncthreads();
}

__global__ __launch_bounds__(256) void fused_mlp_kernel(
    const float* __restrict__ x, const float* __restrict__ b1,
    const __bf16* __restrict__ w1b, const __bf16* __restrict__ w2b,
    const float* __restrict__ probs, float* __restrict__ Mbuf) {
  __shared__ __bf16 xs[16 * LDSW];
  __shared__ __bf16 fs[16 * LDSW];
  const int tid = threadIdx.x, lane = tid & 31, wid = tid >> 5;
  const int nn = lane & 15, lh = lane >> 4;
  const size_t row0 = (size_t)blockIdx.x * 16;
  // stage x tile (16 x 256) into LDS as bf16
#pragma unroll
  for (int it = 0; it < 16; ++it)
    xs[it * LDSW + tid] = f2bf(x[(row0 + it) * CIN + tid]);
  __syncthreads();

  v8f lo0 = {}, lo1 = {}, hi0 = {}, hi1 = {};
  do_chunk(0, b1, w1b, w2b, xs, fs, lane, wid, nn, lh, lo0, lo1);   // h <  256
  for (int hc = 1; hc < 8; ++hc)
    do_chunk(hc, b1, w1b, w2b, xs, fs, lane, wid, nn, lh, hi0, hi1); // h >= 256

  const int b = blockIdx.x / TILES_PER_B;
  const float p0 = probs[b * 2 + 0], p1 = probs[b * 2 + 1];
  const float wlo = p0 + p1, whi = p0;
  const int o0 = (wid * 2 + 0) * 16 + nn;
  const int o1 = (wid * 2 + 1) * 16 + nn;
#pragma unroll
  for (int vv = 0; vv < 8; ++vv) {
    const size_t r = row0 + (size_t)(vv + lh * 8);
    Mbuf[r * COUT + o0] = wlo * lo0[vv] + whi * hi0[vv];
    Mbuf[r * COUT + o1] = wlo * lo1[vv] + whi * hi1[vv];
  }
}

// ----------------- depthwise conv (K=39, same-pad) + residual --------------
__global__ __launch_bounds__(256) void conv_kernel(
    const float* __restrict__ M, const float* __restrict__ cwg,
    float* __restrict__ out) {
  __shared__ float win[(16 + 2 * PAD) * COUT];   // 54 x 256 floats = 54 KB
  const int o = threadIdx.x;
  const int b = blockIdx.y;
  const int t0 = blockIdx.x * 16;
  float cw[KLEN];
#pragma unroll
  for (int j = 0; j < KLEN; ++j) cw[j] = cwg[o * KLEN + j];
  const float* Mb = M + (size_t)b * T_SZ * COUT;
  for (int row = 0; row < 16 + 2 * PAD; ++row) {
    const int t = t0 + row - PAD;
    win[row * COUT + o] = (t >= 0 && t < T_SZ) ? Mb[(size_t)t * COUT + o] : 0.f;
  }
  __syncthreads();
  float* ob = out + (size_t)b * T_SZ * COUT;
#pragma unroll
  for (int tt = 0; tt < 16; ++tt) {
    float acc = win[(PAD + tt) * COUT + o];      // residual xp
#pragma unroll
    for (int j = 0; j < KLEN; ++j) acc += cw[j] * win[(tt + j) * COUT + o];
    ob[(size_t)(t0 + tt) * COUT + o] = acc;
  }
}

// ---------------------------------------------------------------------------
extern "C" void kernel_launch(void* const* d_in, const int* in_sizes, int n_in,
                              void* d_out, int out_size, void* d_ws, size_t ws_size,
                              hipStream_t stream) {
  (void)in_sizes; (void)n_in; (void)out_size; (void)ws_size;
  const float* x      = (const float*)d_in[0];
  const float* w1     = (const float*)d_in[1];
  const float* b1     = (const float*)d_in[2];
  const float* w2     = (const float*)d_in[3];
  const float* conv_w = (const float*)d_in[4];
  const float* gate_w = (const float*)d_in[5];
  const float* gate_b = (const float*)d_in[6];
  const float* tau    = (const float*)d_in[7];
  float* out = (float*)d_out;

  char* ws = (char*)d_ws;
  float*  probs = (float*)(ws);                                    // 64 B
  float*  Mbuf  = (float*)(ws + 256);                              // 32,768,000 B
  __bf16* w1b   = (__bf16*)(ws + 256 + 32768000ULL);               // 1,048,576 B
  __bf16* w2b   = (__bf16*)(ws + 256 + 32768000ULL + 1048576ULL);  // 1,048,576 B

  gate_kernel<<<B_SZ, 256, 0, stream>>>(x, gate_w, gate_b, tau, probs);
  loss_kernel<<<1, 32, 0, stream>>>(probs, out + OUT_N);
  pack_w1_kernel<<<(HID * CIN) / 256, 256, 0, stream>>>(w1, w1b);
  pack_w2_kernel<<<(COUT * HID) / 256, 256, 0, stream>>>(w2, w2b);
  fused_mlp_kernel<<<TILES, 256, 0, stream>>>(x, b1, w1b, w2b, probs, Mbuf);
  conv_kernel<<<dim3(TILES_PER_B, B_SZ), 256, 0, stream>>>(Mbuf, conv_w, out);
}
